// ProtoGated_824633721279
// MI455X (gfx1250) — matrix operand order repaired and checked by reading, not compiled, not run on hardware
//
#include <hip/hip_runtime.h>
#include <math.h>

// ---------------------------------------------------------------------------
// Types for CDNA5 WMMA (gfx1250, wave32)
// ---------------------------------------------------------------------------
typedef __attribute__((ext_vector_type(16))) __bf16 v16bf;
typedef __attribute__((ext_vector_type(8)))  float  v8f;

#define HIDF   128
#define PROTOF 64
#define OUTC   40
#define BN_EPS 1e-5f

__device__ __forceinline__ unsigned short f2bf_bits(float f) {
    unsigned int u = __float_as_uint(f);
    u += 0x7FFFu + ((u >> 16) & 1u);          // round-to-nearest-even
    return (unsigned short)(u >> 16);
}

// pack two floats into two bf16 (round-to-nearest, ties-up) with one v_perm_b32
__device__ __forceinline__ unsigned int pack2bf(float lo, float hi) {
    unsigned int a = __float_as_uint(lo) + 0x8000u;
    unsigned int b = __float_as_uint(hi) + 0x8000u;
    // result bytes (LE): {a[2],a[3],b[2],b[3]} -> selector 0x07060302 with src0=b, src1=a
    return __builtin_amdgcn_perm(b, a, 0x07060302u);
}

union BFrag { v16bf v; unsigned int u[8]; };

// ---------------------------------------------------------------------------
// Utility kernels
// ---------------------------------------------------------------------------
__global__ void k_zero_f32(float* __restrict__ p, long long n) {
    long long i = (long long)blockIdx.x * blockDim.x + threadIdx.x;
    if (i < n) p[i] = 0.0f;
}

__global__ void k_copy_f32(const float* __restrict__ s, float* __restrict__ d, long long n) {
    long long i = (long long)blockIdx.x * blockDim.x + threadIdx.x;
    if (i < n) d[i] = s[i];
}

__global__ void k_cvt_bf16(const float* __restrict__ s, unsigned short* __restrict__ d, int n) {
    int i = blockIdx.x * blockDim.x + threadIdx.x;
    if (i < n) d[i] = f2bf_bits(s[i]);
}

// W_cls [40,256] -> padded bf16 [48,256] (zeros beyond row 39)
__global__ void k_pad_cls(const float* __restrict__ w, unsigned short* __restrict__ d) {
    int i = blockIdx.x * blockDim.x + threadIdx.x;
    if (i >= 48 * 256) return;
    int r = i >> 8;
    d[i] = (r < OUTC) ? f2bf_bits(w[i]) : (unsigned short)0;
}

// ---------------------------------------------------------------------------
// Graph degree / segment-mean scatter
// ---------------------------------------------------------------------------
__global__ void k_deg(const int* __restrict__ tgt, float* __restrict__ deg, int E) {
    int e = blockIdx.x * blockDim.x + threadIdx.x;
    if (e < E) atomicAdd(&deg[tgt[e]], 1.0f);
}

__global__ void k_deg_inv(const float* __restrict__ deg, float* __restrict__ inv, int n) {
    int i = blockIdx.x * blockDim.x + threadIdx.x;
    if (i < n) inv[i] = 1.0f / fmaxf(deg[i], 1.0f);
}

// one wave per edge: gather h[src] (float4/lane), atomic-add into agg[tgt]
__global__ __launch_bounds__(256)
void k_edge_agg(const float* __restrict__ h, const int* __restrict__ src,
                const int* __restrict__ tgt, float* __restrict__ agg, int E) {
    int w    = (blockIdx.x * blockDim.x + threadIdx.x) >> 5;
    int lane = threadIdx.x & 31;
    if (w >= E) return;
    int s = src[w], t = tgt[w];
    const float4 v = *(const float4*)(h + (size_t)s * HIDF + lane * 4);
    float* d = agg + (size_t)t * HIDF + lane * 4;
    atomicAdd(d + 0, v.x); atomicAdd(d + 1, v.y);
    atomicAdd(d + 2, v.z); atomicAdd(d + 3, v.w);
}

// ---------------------------------------------------------------------------
// BatchNorm (training-mode batch stats, biased variance)
// ---------------------------------------------------------------------------
__global__ __launch_bounds__(128)
void k_bn_stats(const float* __restrict__ h, float* __restrict__ stats, int n) {
    int f  = threadIdx.x;                 // 128 features
    int r0 = blockIdx.x * 256;
    int r1 = min(r0 + 256, n);
    float s = 0.0f, sq = 0.0f;
    for (int r = r0; r < r1; ++r) {
        float v = h[(size_t)r * HIDF + f];
        s += v; sq += v * v;
    }
    atomicAdd(&stats[f], s);
    atomicAdd(&stats[HIDF + f], sq);
}

// act: 0 none, 1 relu, 2 sigmoid  (in-place)
__global__ void k_bn_apply(float* __restrict__ h, const float* __restrict__ stats,
                           const float* __restrict__ gamma, const float* __restrict__ beta,
                           int n, int act) {
    long long i = (long long)blockIdx.x * blockDim.x + threadIdx.x;
    long long tot = (long long)n * HIDF;
    if (i >= tot) return;
    int f = (int)(i & (HIDF - 1));
    float invn = 1.0f / (float)n;
    float mu  = stats[f] * invn;
    float var = stats[HIDF + f] * invn - mu * mu;
    float v = (h[i] - mu) * rsqrtf(var + BN_EPS) * gamma[f] + beta[f];
    if (act == 1) v = fmaxf(v, 0.0f);
    else if (act == 2) v = 1.0f / (1.0f + expf(-v));
    h[i] = v;
}

// ---------------------------------------------------------------------------
// WMMA GEMM: out[N,128] = A@W0^T (+ A2@W1^T) + bias
//   A rows optionally scaled by scaleA[row] (segment-mean divide, fused)
//   K in {64,128}; OUTF fixed 128. One wave per 16-row group.
// ---------------------------------------------------------------------------
template<int K, bool HAS_A2, bool HAS_SCALE>
__global__ __launch_bounds__(128)
void k_gemm_dual(const float* __restrict__ A, const float* __restrict__ scaleA,
                 const float* __restrict__ A2,
                 const unsigned short* __restrict__ W0,
                 const unsigned short* __restrict__ W1,
                 const float* __restrict__ bias,
                 float* __restrict__ out, int nGroups) {
    constexpr int NC = K / 32;
    int wid = (blockIdx.x * blockDim.x + threadIdx.x) >> 5;
    if (wid >= nGroups) return;
    int lane = threadIdx.x & 31;
    int m = lane & 15, hh = lane >> 4;
    int row = wid * 16 + m;

    const float* arow = A + (size_t)row * K;
    float sc = 1.0f;
    if (HAS_SCALE) sc = scaleA[row];

    BFrag af[NC], af2[NC];
#pragma unroll
    for (int c = 0; c < NC; ++c) {
        int k0 = c * 32 + hh * 16;
        float4 f0 = *(const float4*)(arow + k0);
        float4 f1 = *(const float4*)(arow + k0 + 4);
        float4 f2 = *(const float4*)(arow + k0 + 8);
        float4 f3 = *(const float4*)(arow + k0 + 12);
        float t[16] = { f0.x, f0.y, f0.z, f0.w, f1.x, f1.y, f1.z, f1.w,
                        f2.x, f2.y, f2.z, f2.w, f3.x, f3.y, f3.z, f3.w };
#pragma unroll
        for (int e = 0; e < 8; ++e)
            af[c].u[e] = pack2bf(t[2 * e] * sc, t[2 * e + 1] * sc);
        if (HAS_A2) {
            const float* a2row = A2 + (size_t)row * K;
            float4 g0 = *(const float4*)(a2row + k0);
            float4 g1 = *(const float4*)(a2row + k0 + 4);
            float4 g2 = *(const float4*)(a2row + k0 + 8);
            float4 g3 = *(const float4*)(a2row + k0 + 12);
            float u[16] = { g0.x, g0.y, g0.z, g0.w, g1.x, g1.y, g1.z, g1.w,
                            g2.x, g2.y, g2.z, g2.w, g3.x, g3.y, g3.z, g3.w };
#pragma unroll
            for (int e = 0; e < 8; ++e)
                af2[c].u[e] = pack2bf(u[2 * e], u[2 * e + 1]);
        }
    }

#pragma unroll
    for (int tile = 0; tile < 8; ++tile) {      // 8 x 16 = 128 output features
        v8f acc = {};
#pragma unroll
        for (int c = 0; c < NC; ++c) {
            v16bf b0 = *(const v16bf*)(W0 + (size_t)(tile * 16 + m) * K + c * 32 + hh * 16);
            acc = __builtin_amdgcn_wmma_f32_16x16x32_bf16(false, af[c].v, false, b0,
                                                          (short)0, acc, false, false);
            if (HAS_A2) {
                v16bf b1 = *(const v16bf*)(W1 + (size_t)(tile * 16 + m) * K + c * 32 + hh * 16);
                acc = __builtin_amdgcn_wmma_f32_16x16x32_bf16(false, af2[c].v, false, b1,
                                                              (short)0, acc, false, false);
            }
        }
        float bb = bias ? bias[tile * 16 + m] : 0.0f;
        float* ob = out + (size_t)(wid * 16 + hh * 8) * HIDF + tile * 16 + m;
#pragma unroll
        for (int r = 0; r < 8; ++r) ob[(size_t)r * HIDF] = acc[r] + bb;
    }
}

// Classifier: logits[N,40] = relu([Hg|Hp]) @ Wcls^T + bcls   (Wcls padded to [48,256])
__global__ __launch_bounds__(128)
void k_gemm_cls(const float* __restrict__ Hg, const float* __restrict__ Hp,
                const unsigned short* __restrict__ Wb, const float* __restrict__ bcls,
                float* __restrict__ logits, int nGroups) {
    int wid = (blockIdx.x * blockDim.x + threadIdx.x) >> 5;
    if (wid >= nGroups) return;
    int lane = threadIdx.x & 31;
    int m = lane & 15, hh = lane >> 4;
    int row = wid * 16 + m;

    BFrag af[8];
#pragma unroll
    for (int c = 0; c < 8; ++c) {
        const float* base = (c < 4) ? (Hg + (size_t)row * HIDF)
                                    : (Hp + (size_t)row * HIDF);
        int k0 = (c & 3) * 32 + hh * 16;
        float4 f0 = *(const float4*)(base + k0);
        float4 f1 = *(const float4*)(base + k0 + 4);
        float4 f2 = *(const float4*)(base + k0 + 8);
        float4 f3 = *(const float4*)(base + k0 + 12);
        float t[16] = { f0.x, f0.y, f0.z, f0.w, f1.x, f1.y, f1.z, f1.w,
                        f2.x, f2.y, f2.z, f2.w, f3.x, f3.y, f3.z, f3.w };
#pragma unroll
        for (int e = 0; e < 8; ++e)
            af[c].u[e] = pack2bf(fmaxf(t[2 * e], 0.0f), fmaxf(t[2 * e + 1], 0.0f));
    }

#pragma unroll
    for (int tile = 0; tile < 3; ++tile) {      // 48 padded output cols
        v8f acc = {};
#pragma unroll
        for (int c = 0; c < 8; ++c) {
            v16bf b = *(const v16bf*)(Wb + (size_t)(tile * 16 + m) * 256 + c * 32 + hh * 16);
            acc = __builtin_amdgcn_wmma_f32_16x16x32_bf16(false, af[c].v, false, b,
                                                          (short)0, acc, false, false);
        }
        int col = tile * 16 + m;
        if (col < OUTC) {
            float bb = bcls[col];
            float* ob = logits + (size_t)(wid * 16 + hh * 8) * OUTC + col;
#pragma unroll
            for (int r = 0; r < 8; ++r) ob[(size_t)r * OUTC] = acc[r] + bb;
        }
    }
}

// ---------------------------------------------------------------------------
// Gate: cosine-sim segment mean with self loops
// ---------------------------------------------------------------------------
__global__ __launch_bounds__(256)
void k_alpha_norm(const float* __restrict__ alpha, float* __restrict__ invn,
                  float* __restrict__ ssq, int n) {
    int w    = (blockIdx.x * blockDim.x + threadIdx.x) >> 5;
    int lane = threadIdx.x & 31;
    if (w >= n) return;
    float2 a = *(const float2*)(alpha + (size_t)w * PROTOF + lane * 2);
    float s = a.x * a.x + a.y * a.y;
#pragma unroll
    for (int o = 16; o > 0; o >>= 1) s += __shfl_xor(s, o, 32);
    if (lane == 0) {
        ssq[w]  = s;
        invn[w] = 1.0f / fmaxf(sqrtf(s), 1e-12f);
    }
}

__global__ __launch_bounds__(256)
void k_gate_edges(const float* __restrict__ alpha, const float* __restrict__ invn,
                  const int* __restrict__ src, const int* __restrict__ tgt,
                  float* __restrict__ gsum, int E) {
    int w    = (blockIdx.x * blockDim.x + threadIdx.x) >> 5;
    int lane = threadIdx.x & 31;
    if (w >= E) return;
    int s = src[w], t = tgt[w];
    float2 as = *(const float2*)(alpha + (size_t)s * PROTOF + lane * 2);
    float2 at = *(const float2*)(alpha + (size_t)t * PROTOF + lane * 2);
    float d = as.x * at.x + as.y * at.y;
#pragma unroll
    for (int o = 16; o > 0; o >>= 1) d += __shfl_xor(d, o, 32);
    if (lane == 0) atomicAdd(&gsum[t], d * invn[s] * invn[t]);
}

__global__ void k_gate_self(float* __restrict__ gsum, const float* __restrict__ ssq,
                            const float* __restrict__ invn, int n) {
    int i = blockIdx.x * blockDim.x + threadIdx.x;
    if (i < n) gsum[i] += ssq[i] * invn[i] * invn[i];
}

__global__ void k_gate_final(const float* __restrict__ gsum, const float* __restrict__ deg,
                             const float* __restrict__ temp, float* __restrict__ og, int n) {
    int i = blockIdx.x * blockDim.x + threadIdx.x;
    if (i >= n) return;
    float cnt = fmaxf(deg[i] + 1.0f, 1.0f);
    float g   = temp[0] * (gsum[i] / cnt);
    og[i] = 1.0f / (1.0f + expf(-g));
}

// ---------------------------------------------------------------------------
// log_softmax over OUTC=40
// ---------------------------------------------------------------------------
__global__ void k_logsoftmax(const float* __restrict__ logits, float* __restrict__ out, int n) {
    int i = blockIdx.x * blockDim.x + threadIdx.x;
    if (i >= n) return;
    const float* l = logits + (size_t)i * OUTC;
    float mx = -3.4e38f;
    for (int j = 0; j < OUTC; ++j) mx = fmaxf(mx, l[j]);
    float s = 0.0f;
    for (int j = 0; j < OUTC; ++j) s += expf(l[j] - mx);
    float lse = mx + logf(s);
    float* o = out + (size_t)i * OUTC;
    for (int j = 0; j < OUTC; ++j) o[j] = l[j] - lse;
}

// ---------------------------------------------------------------------------
// Host orchestration
// ---------------------------------------------------------------------------
static inline long long alignup(long long x, long long a) { return (x + a - 1) / a * a; }

extern "C" void kernel_launch(void* const* d_in, const int* in_sizes, int n_in,
                              void* d_out, int out_size, void* d_ws, size_t ws_size,
                              hipStream_t stream) {
    const float* x      = (const float*)d_in[0];
    const float* alpha  = (const float*)d_in[1];
    const int*   eidx   = (const int*)d_in[2];
    const float* Wl     = (const float*)d_in[3];
    const float* bl     = (const float*)d_in[4];
    const float* Wr     = (const float*)d_in[5];
    const float* sgam   = (const float*)d_in[6];
    const float* sbet   = (const float*)d_in[7];
    const float* W0     = (const float*)d_in[8];
    const float* b0     = (const float*)d_in[9];
    const float* W1     = (const float*)d_in[10];
    const float* b1     = (const float*)d_in[11];
    const float* W2     = (const float*)d_in[12];
    const float* b2     = (const float*)d_in[13];
    const float* mgam   = (const float*)d_in[14];
    const float* mbet   = (const float*)d_in[15];
    const float* Wcls   = (const float*)d_in[16];
    const float* bcls   = (const float*)d_in[17];
    const float* temp   = (const float*)d_in[18];

    const int N = in_sizes[0] / HIDF;
    const int E = in_sizes[2] / 2;
    const int* src = eidx;
    const int* tgt = eidx + E;
    const int nGroups = N / 16;

    // ---- workspace layout ----
    char* ws = (char*)d_ws; long long off = 0;
    auto take = [&](long long bytes) { char* p = ws + off; off = alignup(off + bytes, 256); return p; };
    float* bufA   = (float*)take((long long)N * HIDF * 4);   // ping
    float* bufB   = (float*)take((long long)N * HIDF * 4);   // pong
    float* agg    = (float*)take((long long)N * HIDF * 4);   // agg, later proto ping
    float* logits = (float*)take((long long)N * OUTC * 4);
    float* deg    = (float*)take((long long)N * 4);
    float* deginv = (float*)take((long long)N * 4);
    float* invn   = (float*)take((long long)N * 4);
    float* ssq    = (float*)take((long long)N * 4);
    float* gsum   = (float*)take((long long)N * 4);
    float* stats  = (float*)take(2 * HIDF * 4);
    unsigned short* Wlb   = (unsigned short*)take(3LL * HIDF * HIDF * 2);
    unsigned short* Wrb   = (unsigned short*)take(3LL * HIDF * HIDF * 2);
    unsigned short* W0b   = (unsigned short*)take((long long)HIDF * PROTOF * 2);
    unsigned short* W1b   = (unsigned short*)take((long long)HIDF * HIDF * 2);
    unsigned short* W2b   = (unsigned short*)take((long long)HIDF * HIDF * 2);
    unsigned short* Wclsb = (unsigned short*)take(48LL * 256 * 2);
    (void)ws_size; (void)n_in; (void)out_size;

    const int T = 256;
    long long nh = (long long)N * HIDF;
    int gbNH   = (int)((nh + T - 1) / T);
    int gbN    = (N + T - 1) / T;
    int gbE    = (E + T - 1) / T;
    int gbEw   = (int)(((long long)E * 32 + T - 1) / T);     // wave-per-edge
    int gbNw   = (int)(((long long)N * 32 + T - 1) / T);     // wave-per-node
    int gbGemm = (nGroups * 32 + 127) / 128;                  // 4 waves/block

    // ---- weight conversion to bf16 ----
    k_cvt_bf16<<<(3 * HIDF * HIDF + T - 1) / T, T, 0, stream>>>(Wl, Wlb, 3 * HIDF * HIDF);
    k_cvt_bf16<<<(3 * HIDF * HIDF + T - 1) / T, T, 0, stream>>>(Wr, Wrb, 3 * HIDF * HIDF);
    k_cvt_bf16<<<(HIDF * PROTOF + T - 1) / T, T, 0, stream>>>(W0, W0b, HIDF * PROTOF);
    k_cvt_bf16<<<(HIDF * HIDF + T - 1) / T, T, 0, stream>>>(W1, W1b, HIDF * HIDF);
    k_cvt_bf16<<<(HIDF * HIDF + T - 1) / T, T, 0, stream>>>(W2, W2b, HIDF * HIDF);
    k_pad_cls<<<(48 * 256 + T - 1) / T, T, 0, stream>>>(Wcls, Wclsb);

    // ---- degrees ----
    k_zero_f32<<<gbN, T, 0, stream>>>(deg, N);
    k_deg<<<gbE, T, 0, stream>>>(tgt, deg, E);
    k_deg_inv<<<gbN, T, 0, stream>>>(deg, deginv, N);

    // ---- SAGE branch ----
    k_copy_f32<<<gbNH, T, 0, stream>>>(x, bufA, nh);
    float* cur = bufA; float* nxt = bufB;
    for (int i = 0; i < 3; ++i) {
        k_zero_f32<<<gbNH, T, 0, stream>>>(agg, nh);
        k_edge_agg<<<gbEw, T, 0, stream>>>(cur, src, tgt, agg, E);
        k_gemm_dual<HIDF, true, true><<<gbGemm, 128, 0, stream>>>(
            agg, deginv, cur, Wlb + (size_t)i * HIDF * HIDF, Wrb + (size_t)i * HIDF * HIDF,
            bl + i * HIDF, nxt, nGroups);
        if (i < 2) {
            k_zero_f32<<<1, 256, 0, stream>>>(stats, 2 * HIDF);
            k_bn_stats<<<(N + 255) / 256, 128, 0, stream>>>(nxt, stats, N);
            k_bn_apply<<<gbNH, T, 0, stream>>>(nxt, stats, sgam + i * HIDF, sbet + i * HIDF, N, 1);
        }
        float* t2 = cur; cur = nxt; nxt = t2;
    }
    float* Hgnn = cur;          // after 3 swaps; `nxt` buffer is now free

    // ---- Proto MLP branch (reuse agg + free ping buffer) ----
    float* pA = agg;
    float* pB = nxt;
    k_gemm_dual<PROTOF, false, false><<<gbGemm, 128, 0, stream>>>(
        alpha, nullptr, nullptr, W0b, nullptr, b0, pA, nGroups);
    k_zero_f32<<<1, 256, 0, stream>>>(stats, 2 * HIDF);
    k_bn_stats<<<(N + 255) / 256, 128, 0, stream>>>(pA, stats, N);
    k_bn_apply<<<gbNH, T, 0, stream>>>(pA, stats, mgam, mbet, N, 2);

    k_gemm_dual<HIDF, false, false><<<gbGemm, 128, 0, stream>>>(
        pA, nullptr, nullptr, W1b, nullptr, b1, pB, nGroups);
    k_zero_f32<<<1, 256, 0, stream>>>(stats, 2 * HIDF);
    k_bn_stats<<<(N + 255) / 256, 128, 0, stream>>>(pB, stats, N);
    k_bn_apply<<<gbNH, T, 0, stream>>>(pB, stats, mgam + HIDF, mbet + HIDF, N, 2);

    k_gemm_dual<HIDF, false, false><<<gbGemm, 128, 0, stream>>>(
        pB, nullptr, nullptr, W2b, nullptr, b2, pA, nGroups);
    float* Hproto = pA;

    // ---- classifier + log_softmax ----
    k_gemm_cls<<<gbGemm, 128, 0, stream>>>(Hgnn, Hproto, Wclsb, bcls, logits, nGroups);
    k_logsoftmax<<<gbN, T, 0, stream>>>(logits, (float*)d_out, N);

    // ---- homophily gate ----
    k_zero_f32<<<gbN, T, 0, stream>>>(gsum, N);
    k_alpha_norm<<<gbNw, T, 0, stream>>>(alpha, invn, ssq, N);
    k_gate_edges<<<gbEw, T, 0, stream>>>(alpha, invn, src, tgt, gsum, E);
    k_gate_self<<<gbN, T, 0, stream>>>(gsum, ssq, invn, N);
    k_gate_final<<<gbN, T, 0, stream>>>(gsum, deg, temp, (float*)d_out + (size_t)N * OUTC, N);
}
